// SimpleGAT_38362647888480
// MI455X (gfx1250) — matrix operand order repaired and checked by reading, not compiled, not run on hardware
//
#include <hip/hip_runtime.h>
#include <hip/hip_bf16.h>
#include <math.h>

#define NSK   50000
#define BATCH 1024
#define EDG   16384
#define DINV  1536
#define HEADS 8
#define HDV   12288

// ordered-monotone encoding of -inf: ~as_uint(-inf) = ~0xFF800000 = 0x007FFFFF
#define ENC_NEG_INF 0x007FFFFFu

typedef __attribute__((ext_vector_type(16))) __bf16 v16bf;
typedef __attribute__((ext_vector_type(8)))  float  v8f;
typedef unsigned short ush;

union Frag { unsigned int u[8]; v16bf v; };

__device__ __forceinline__ ush f2bf(float f) {
  unsigned int u = __float_as_uint(f);
  u += 0x7FFFu + ((u >> 16) & 1u);   // round-to-nearest-even
  return (ush)(u >> 16);
}

__device__ __forceinline__ unsigned int encOrd(float f) {
  unsigned int u = __float_as_uint(f);
  return (u & 0x80000000u) ? ~u : (u | 0x80000000u);
}
__device__ __forceinline__ float decOrd(unsigned int e) {
  unsigned int u = (e & 0x80000000u) ? (e & 0x7FFFFFFFu) : ~e;
  return __uint_as_float(u);
}

// ---------------- generic fills ----------------
__global__ void fill_f32(float* p, long n, float v) {
  long i = (long)blockIdx.x * blockDim.x + threadIdx.x;
  if (i < n) p[i] = v;
}
__global__ void fill_u32(unsigned int* p, long n, unsigned int v) {
  long i = (long)blockIdx.x * blockDim.x + threadIdx.x;
  if (i < n) p[i] = v;
}

// ---------------- GEMM helpers ----------------
// Stage one 128x32 A tile and 32x128 W tile (fp32 -> bf16, W transposed) into LDS.
// EDGE=false & !kt: fully unguarded fast path (128B coalesced loads + prefetch).
template<bool EDGE>
__device__ __forceinline__ void stage_tile(const float* __restrict__ A, int lda,
                                           const float* __restrict__ W, int ldw,
                                           int m0, int n0, int k0, int K, int N, bool kt,
                                           int tid, ush (*As)[34], ush (*Bs)[34]) {
  if (!EDGE && !kt) {
#pragma unroll
    for (int t = 0; t < 4; ++t) {
      int i = tid + t * 256;
      int r = i >> 3;                      // 0..127
      int c = (i & 7) * 4;                 // 0..28
      float4 v = *(const float4*)(A + (size_t)(m0 + r) * lda + (k0 + c));
      As[r][c + 0] = f2bf(v.x); As[r][c + 1] = f2bf(v.y);
      As[r][c + 2] = f2bf(v.z); As[r][c + 3] = f2bf(v.w);
    }
#pragma unroll
    for (int t = 0; t < 4; ++t) {
      int i  = tid + t * 256;
      int kr = i >> 5;                     // 0..31
      int c  = (i & 31) * 4;               // 0..124
      float4 v = *(const float4*)(W + (size_t)(k0 + kr) * ldw + (n0 + c));
      Bs[c + 0][kr] = f2bf(v.x); Bs[c + 1][kr] = f2bf(v.y);
      Bs[c + 2][kr] = f2bf(v.z); Bs[c + 3][kr] = f2bf(v.w);
    }
    if (k0 + 32 < K) {  // hint next K-tile (global_prefetch_b8; speculative, drop on fault)
      __builtin_prefetch(A + (size_t)(m0 + (tid >> 3)) * lda + (k0 + 32), 0, 1);
      __builtin_prefetch(W + (size_t)(k0 + 32 + (tid >> 5)) * ldw + n0 + ((tid & 31) * 4), 0, 1);
    }
  } else {
#pragma unroll
    for (int t = 0; t < 4; ++t) {
      int i = tid + t * 256;
      int r = i >> 3;
      int c = (i & 7) * 4;
      const float* ap = A + (size_t)(m0 + r) * lda + (k0 + c);
      float4 v = {0.f, 0.f, 0.f, 0.f};
      if (k0 + c + 3 < K) {
        v = *(const float4*)ap;
      } else {
        if (k0 + c + 0 < K) v.x = ap[0];
        if (k0 + c + 1 < K) v.y = ap[1];
        if (k0 + c + 2 < K) v.z = ap[2];
        if (k0 + c + 3 < K) v.w = ap[3];
      }
      As[r][c + 0] = f2bf(v.x); As[r][c + 1] = f2bf(v.y);
      As[r][c + 2] = f2bf(v.z); As[r][c + 3] = f2bf(v.w);
    }
#pragma unroll
    for (int t = 0; t < 4; ++t) {
      int i  = tid + t * 256;
      int kr = i >> 5;
      int c  = (i & 31) * 4;
      int kg = k0 + kr;
      float4 v = {0.f, 0.f, 0.f, 0.f};
      if (kg < K) {
        const float* wp = W + (size_t)kg * ldw + (n0 + c);
        if (n0 + c + 3 < N) {
          v = *(const float4*)wp;
        } else {
          if (n0 + c + 0 < N) v.x = wp[0];
          if (n0 + c + 1 < N) v.y = wp[1];
          if (n0 + c + 2 < N) v.z = wp[2];
          if (n0 + c + 3 < N) v.w = wp[3];
        }
      }
      Bs[c + 0][kr] = f2bf(v.x); Bs[c + 1][kr] = f2bf(v.y);
      Bs[c + 2][kr] = f2bf(v.z); Bs[c + 3][kr] = f2bf(v.w);
    }
  }
}

// 8 WMMAs: 2 (M) x 4 (N) 16x16 tiles per wave, one 32-deep K step.
__device__ __forceinline__ void mma_step(const ush (*As)[34], const ush (*Bs)[34],
                                         v8f acc[2][4], int wm, int wn, int kgrp, int l16) {
  Frag af[2], bfr[4];
#pragma unroll
  for (int r = 0; r < 8; ++r) {
    int ka = ((r >> 2) << 4) + ((r & 3) << 1) + (kgrp << 3);   // A 16-bit layout
    af[0].u[r] = *(const unsigned int*)&As[wm * 32 + l16][ka];
    af[1].u[r] = *(const unsigned int*)&As[wm * 32 + 16 + l16][ka];
    int kb = ((r >> 2) << 4) + ((r & 3) << 1) + (kgrp << 3);   // B 16-bit layout
    bfr[0].u[r] = *(const unsigned int*)&Bs[wn * 64 +  0 + l16][kb];
    bfr[1].u[r] = *(const unsigned int*)&Bs[wn * 64 + 16 + l16][kb];
    bfr[2].u[r] = *(const unsigned int*)&Bs[wn * 64 + 32 + l16][kb];
    bfr[3].u[r] = *(const unsigned int*)&Bs[wn * 64 + 48 + l16][kb];
  }
#pragma unroll
  for (int tm = 0; tm < 2; ++tm)
#pragma unroll
    for (int tn = 0; tn < 4; ++tn)
      acc[tm][tn] = __builtin_amdgcn_wmma_f32_16x16x32_bf16(
          false, af[tm].v, false, bfr[tn].v, (short)0, acc[tm][tn], false, false);
}

// ---------------- bf16 WMMA GEMM: C[M,N] = act(A[M,K]@W[K,N] + bias + Cadd) ----
// 128x128 block tile, 8 waves (4x2), double-buffered LDS, one barrier per K-step.
__global__ __launch_bounds__(256)
void gemm_bf16_wmma(const float* __restrict__ A, int lda,
                    const float* __restrict__ W, int ldw,
                    const float* __restrict__ bias,
                    const float* __restrict__ Cadd, int ldcadd,
                    float* __restrict__ C, int ldc,
                    int N, int K, int act) {
  __shared__ ush As[2][128][34];
  __shared__ ush Bs[2][128][34];

  const int tid  = threadIdx.x;
  const int wave = tid >> 5;
  const int lane = tid & 31;
  const int wm   = wave >> 1;              // 0..3  -> 32 rows each
  const int wn   = wave & 1;               // 0..1  -> 64 cols each
  const int m0   = blockIdx.y * 128;
  const int n0   = blockIdx.x * 128;
  const int kgrp = lane >> 4;
  const int l16  = lane & 15;

  v8f acc[2][4] = {};

  const int  nsteps  = (K + 31) >> 5;
  const bool hasTail = (K & 31) != 0;
  const bool edgeN   = (n0 + 128 > N);

  if (!edgeN) {
    stage_tile<false>(A, lda, W, ldw, m0, n0, 0, K, N,
                      hasTail && nsteps == 1, tid, As[0], Bs[0]);
    int p = 0;
    for (int s = 1; s < nsteps; ++s) {
      __syncthreads();
      bool kt = hasTail && (s == nsteps - 1);
      stage_tile<false>(A, lda, W, ldw, m0, n0, s * 32, K, N, kt, tid, As[p ^ 1], Bs[p ^ 1]);
      mma_step(As[p], Bs[p], acc, wm, wn, kgrp, l16);
      p ^= 1;
    }
    __syncthreads();
    mma_step(As[p], Bs[p], acc, wm, wn, kgrp, l16);
  } else {
    stage_tile<true>(A, lda, W, ldw, m0, n0, 0, K, N, true, tid, As[0], Bs[0]);
    int p = 0;
    for (int s = 1; s < nsteps; ++s) {
      __syncthreads();
      stage_tile<true>(A, lda, W, ldw, m0, n0, s * 32, K, N, true, tid, As[p ^ 1], Bs[p ^ 1]);
      mma_step(As[p], Bs[p], acc, wm, wn, kgrp, l16);
      p ^= 1;
    }
    __syncthreads();
    mma_step(As[p], Bs[p], acc, wm, wn, kgrp, l16);
  }

  // ---- epilogue: lane holds (n = lane&15, m = r + 8*(lane>>4)) per C layout ----
#pragma unroll
  for (int tm = 0; tm < 2; ++tm) {
#pragma unroll
    for (int tn = 0; tn < 4; ++tn) {
      int col = n0 + wn * 64 + tn * 16 + l16;
      if (col < N) {
        float bv = bias ? bias[col] : 0.f;
#pragma unroll
        for (int r = 0; r < 8; ++r) {
          int row = m0 + wm * 32 + tm * 16 + r + 8 * kgrp;
          float v = acc[tm][tn][r] + bv;
          if (Cadd) v += Cadd[(size_t)row * ldcadd + col];
          if (act)  v = v > 0.f ? v : 0.01f * v;
          C[(size_t)row * ldc + col] = v;
        }
      }
    }
  }
}

// ---------------- attention vector precompute: v[din,h] = sum_d W[din,h*D+d]*att[h,d]
__global__ void att_reduce(const float* __restrict__ Wsrc, const float* __restrict__ Wdst,
                           const float* __restrict__ as_att, const float* __restrict__ ad_att,
                           float* __restrict__ vsrc, float* __restrict__ vdst) {
  int i = blockIdx.x * blockDim.x + threadIdx.x;   // din*8+h
  if (i >= DINV * HEADS) return;
  int din = i >> 3, h = i & 7;
  const float* w1 = Wsrc + (size_t)din * HDV + h * DINV;
  const float* w2 = Wdst + (size_t)din * HDV + h * DINV;
  const float* a1 = as_att + h * DINV;
  const float* a2 = ad_att + h * DINV;
  float s1 = 0.f, s2 = 0.f;
  for (int d = 0; d < DINV; ++d) { s1 += w1[d] * a1[d]; s2 += w2[d] * a2[d]; }
  vsrc[i] = s1; vdst[i] = s2;
}

// ---------------- a_d[b,h] = x_resume[b] . vdst[:,h]
__global__ void ad_kernel(const float* __restrict__ x_resume, const float* __restrict__ vdst,
                          float* __restrict__ ad) {
  int i = blockIdx.x * blockDim.x + threadIdx.x;   // b*8+h
  if (i >= BATCH * HEADS) return;
  int b = i >> 3, h = i & 7;
  const float* xr = x_resume + (size_t)b * DINV;
  float s = 0.f;
  for (int d = 0; d < DINV; ++d) s += xr[d] * vdst[d * 8 + h];
  ad[i] = s;
}

// ---------------- edge logits: e = lrelu0.2(a_s[src]+a_d[dst]); seg-max via ordered atomicMax
__global__ void edge_logits(const float* __restrict__ x_skill, const float* __restrict__ vsrc,
                            const float* __restrict__ ad, const int* __restrict__ src,
                            const int* __restrict__ dst, float* __restrict__ eatt,
                            unsigned int* __restrict__ mEnc) {
  int i = blockIdx.x * blockDim.x + threadIdx.x;   // e*8+h
  if (i >= EDG * HEADS) return;
  int e = i >> 3, h = i & 7;
  int s = src[e], d = dst[e];
  const float* xr = x_skill + (size_t)s * DINV;
  float a = 0.f;
  for (int din = 0; din < DINV; ++din) a += xr[din] * vsrc[din * 8 + h];
  float v = a + ad[d * 8 + h];
  v = v > 0.f ? v : 0.2f * v;
  eatt[i] = v;
  atomicMax(&mEnc[d * 8 + h], encOrd(v));
}

// ---------------- ex = exp(e - m); seg-sum denom
__global__ void seg_exp(const float* __restrict__ eatt, const unsigned int* __restrict__ mEnc,
                        const int* __restrict__ dst, float* __restrict__ ex,
                        float* __restrict__ denom) {
  int i = blockIdx.x * blockDim.x + threadIdx.x;
  if (i >= EDG * HEADS) return;
  int e = i >> 3, h = i & 7;
  float m = decOrd(mEnc[dst[e] * 8 + h]);
  if (!isfinite(m)) m = 0.f;
  float v = expf(eatt[i] - m);
  ex[i] = v;
  atomicAdd(&denom[dst[e] * 8 + h], v);
}

// ---------------- alpha = ex / (denom + 1e-16)
__global__ void alpha_write(const float* __restrict__ ex, const float* __restrict__ denom,
                            const int* __restrict__ dst, float* __restrict__ alpha) {
  int i = blockIdx.x * blockDim.x + threadIdx.x;
  if (i >= EDG * HEADS) return;
  int e = i >> 3, h = i & 7;
  alpha[i] = ex[i] / (denom[dst[e] * 8 + h] + 1e-16f);
}

// ---------------- Y[(dst*8+h), din] += alpha[e,h] * x_skill[src, din]
__global__ void scatter_y(const float* __restrict__ x_skill, const int* __restrict__ src,
                          const int* __restrict__ dst, const float* __restrict__ alpha,
                          float* __restrict__ y) {
  int e = blockIdx.x;
  int s = src[e], d = dst[e];
  __shared__ float al[HEADS];
  if (threadIdx.x < HEADS) al[threadIdx.x] = alpha[(size_t)e * HEADS + threadIdx.x];
  __syncthreads();
  const float* xr = x_skill + (size_t)s * DINV;
  for (int i = threadIdx.x; i < DINV; i += blockDim.x) {
    float xv = xr[i];
#pragma unroll
    for (int h = 0; h < HEADS; ++h)
      atomicAdd(&y[((size_t)d * HEADS + h) * DINV + i], al[h] * xv);
  }
}

// ---------------- agg[dst, din] += x_skill[src, din]
__global__ void scatter_agg(const float* __restrict__ x_skill, const int* __restrict__ src,
                            const int* __restrict__ dst, float* __restrict__ agg) {
  int e = blockIdx.x;
  int s = src[e], d = dst[e];
  const float* xr = x_skill + (size_t)s * DINV;
  float* ar = agg + (size_t)d * DINV;
  for (int i = threadIdx.x; i < DINV; i += blockDim.x)
    atomicAdd(&ar[i], xr[i]);
}

// ---------------- final layer: score[b] = x[b,:64] . w + b0
__global__ void final_dot(const float* __restrict__ x, const float* __restrict__ w,
                          const float* __restrict__ b, float* __restrict__ out) {
  int bi = blockIdx.x * blockDim.x + threadIdx.x;
  if (bi >= BATCH) return;
  const float* xr = x + (size_t)bi * 64;
  float s = b[0];
#pragma unroll
  for (int i = 0; i < 64; ++i) s += xr[i] * w[i];
  out[bi] = s;
}

static inline void launch_gemm(const float* A, int lda, const float* W, int ldw,
                               const float* bias, const float* Cadd, int ldcadd,
                               float* C, int ldc, int M, int N, int K, int act,
                               hipStream_t s) {
  dim3 grid((N + 127) / 128, M / 128);
  gemm_bf16_wmma<<<grid, 256, 0, s>>>(A, lda, W, ldw, bias, Cadd, ldcadd, C, ldc, N, K, act);
}

extern "C" void kernel_launch(void* const* d_in, const int* in_sizes, int n_in,
                              void* d_out, int out_size, void* d_ws, size_t ws_size,
                              hipStream_t stream) {
  const float* x_skill   = (const float*)d_in[0];
  const float* x_job     = (const float*)d_in[1];
  const float* x_resume  = (const float*)d_in[2];
  const int*   ei_job_src = (const int*)d_in[3];
  const int*   ei_job_dst = (const int*)d_in[4];
  const int*   ei_res_src = (const int*)d_in[5];
  const int*   ei_res_dst = (const int*)d_in[6];
  const float* gc_w_rel  = (const float*)d_in[7];
  const float* gc_b_rel  = (const float*)d_in[8];
  const float* gc_w_root = (const float*)d_in[9];
  const float* gat_w_src = (const float*)d_in[10];
  const float* gat_w_dst = (const float*)d_in[11];
  const float* att_src   = (const float*)d_in[12];
  const float* att_dst   = (const float*)d_in[13];
  const float* gat_b     = (const float*)d_in[14];

  float* score = (float*)d_out;              // [1024]
  float* alpha = (float*)d_out + BATCH;      // [E, H]

  // ---- workspace layout (bufB aliases the dead scatter region) ----
  char* base = (char*)d_ws;
  const size_t Y_BYTES    = (size_t)BATCH * HEADS * DINV * 4;   // 50331648
  const size_t AGG_BYTES  = (size_t)BATCH * DINV * 4;           // 6291456
  float* y    = (float*)(base);
  float* agg  = (float*)(base + Y_BYTES);
  float* tmpJ = (float*)(base + Y_BYTES + AGG_BYTES);
  float* bufB = (float*)(base);                                  // reused after y/agg/tmpJ die
  size_t off = Y_BYTES + 2 * AGG_BYTES;                          // 62914560
  float* bufA = (float*)(base + off); off += (size_t)BATCH * 13824 * 4;
  float* vsrc = (float*)(base + off); off += 49152;
  float* vdst = (float*)(base + off); off += 49152;
  float* adb  = (float*)(base + off); off += 32768;
  float* eatt = (float*)(base + off); off += 524288;
  float* exb  = (float*)(base + off); off += 524288;
  float* denom = (float*)(base + off); off += 32768;
  unsigned int* mEnc = (unsigned int*)(base + off); off += 32768;

  // ---- init accumulators ----
  {
    long ny = (long)BATCH * HEADS * DINV;
    fill_f32<<<(unsigned)((ny + 255) / 256), 256, 0, stream>>>(y, ny, 0.f);
    long na = (long)BATCH * DINV;
    fill_f32<<<(unsigned)((na + 255) / 256), 256, 0, stream>>>(agg, na, 0.f);
    fill_f32<<<32, 256, 0, stream>>>(denom, BATCH * HEADS, 0.f);
    fill_u32<<<32, 256, 0, stream>>>(mEnc, BATCH * HEADS, ENC_NEG_INF);
  }

  // ---- GAT attention path (factored) ----
  att_reduce<<<(DINV * HEADS + 255) / 256, 256, 0, stream>>>(gat_w_src, gat_w_dst,
                                                             att_src, att_dst, vsrc, vdst);
  ad_kernel<<<(BATCH * HEADS + 255) / 256, 256, 0, stream>>>(x_resume, vdst, adb);
  edge_logits<<<(EDG * HEADS + 255) / 256, 256, 0, stream>>>(x_skill, vsrc, adb,
                                                             ei_res_src, ei_res_dst, eatt, mEnc);
  seg_exp<<<(EDG * HEADS + 255) / 256, 256, 0, stream>>>(eatt, mEnc, ei_res_dst, exb, denom);
  alpha_write<<<(EDG * HEADS + 255) / 256, 256, 0, stream>>>(exb, denom, ei_res_dst, alpha);
  scatter_y<<<EDG, 256, 0, stream>>>(x_skill, ei_res_src, ei_res_dst, alpha, y);

  // ---- GraphConv scatter ----
  scatter_agg<<<EDG, 256, 0, stream>>>(x_skill, ei_job_src, ei_job_dst, agg);

  // ---- GraphConv GEMMs -> x0[:, 0:1536] (lrelu fused) ----
  launch_gemm(x_job, DINV, gc_w_root, DINV, nullptr, nullptr, 0,
              tmpJ, DINV, BATCH, DINV, DINV, 0, stream);
  launch_gemm(agg, DINV, gc_w_rel, DINV, gc_b_rel, tmpJ, DINV,
              bufA, 13824, BATCH, DINV, DINV, 1, stream);

  // ---- GAT per-head GEMMs -> x0[:, 1536 + h*1536 : ...] (bias + lrelu fused) ----
  for (int h = 0; h < HEADS; ++h) {
    launch_gemm(y + h * DINV, HEADS * DINV,
                gat_w_src + h * DINV, HDV,
                gat_b + h * DINV, nullptr, 0,
                bufA + DINV + h * DINV, 13824,
                BATCH, DINV, DINV, 1, stream);
  }

  // ---- deep MLP (layers 0..8 via WMMA GEMM, layer 9 via dot) ----
  const int mlp_dims[11] = {13824, 12000, 6000, 3072, 2048, 1024, 512, 256, 128, 64, 1};
  float* cur = bufA;
  float* nxt = bufB;
  for (int i = 0; i < 9; ++i) {
    launch_gemm(cur, mlp_dims[i],
                (const float*)d_in[15 + 2 * i], mlp_dims[i + 1],
                (const float*)d_in[16 + 2 * i], nullptr, 0,
                nxt, mlp_dims[i + 1],
                BATCH, mlp_dims[i + 1], mlp_dims[i], 1, stream);
    float* t = cur; cur = nxt; nxt = t;
  }
  final_dot<<<(BATCH + 255) / 256, 256, 0, stream>>>(cur, (const float*)d_in[33],
                                                     (const float*)d_in[34], score);
}
